// RNN_45878840656508
// MI455X (gfx1250) — compile-verified, matrix operand order checked by reading, not run on hardware
//
#include <hip/hip_runtime.h>
#include <hip/hip_bf16.h>

// MI455X / gfx1250: wave32, WMMA 16x16x32 bf16 (f32 accum), async global->LDS.
// Plan:
//   k1: cast x, W_ih, W_hh to bf16 in workspace
//   k2: xp[t][b][h] = (x @ W_ih^T + b_ih + b_hh) via WMMA bf16, stored bf16 [T,B,H]
//   k3: persistent per-batch-tile RNN scan: 16 blocks x 512 threads (16 waves),
//       h (16x512) double-buffered in LDS bf16, xp tile (16KB/step) streamed in
//       with GLOBAL_LOAD_ASYNC_TO_LDS_B128 (ASYNCcnt) overlapped with WMMAs,
//       W_hh bf16 streamed from L2, out[b,t] = h_new . w_ho fused.
// Workspace layout (bytes): [0,16MB) x_bf16 | [16MB,+64KB) Wih_bf16 |
//   [+512KB) Whh_bf16 | [17,367,040 , +128MB) xp_bf16.  Needs ~145MB of ws.

typedef __bf16 bf16_t;
typedef __bf16 v16bf __attribute__((ext_vector_type(16)));
typedef __bf16 v8bf  __attribute__((ext_vector_type(8)));
typedef float  v8f   __attribute__((ext_vector_type(8)));

#define B_ 256
#define T_ 512
#define D_ 64
#define H_ 512

static __device__ __forceinline__ v16bf cat8(v8bf lo, v8bf hi) {
  return __builtin_shufflevector(lo, hi, 0, 1, 2, 3, 4, 5, 6, 7,
                                 8, 9, 10, 11, 12, 13, 14, 15);
}

static __device__ __forceinline__ v8f wmma_bf16(v16bf a, v16bf b, v8f c) {
  return __builtin_amdgcn_wmma_f32_16x16x32_bf16(
      /*neg_a=*/false, a, /*neg_b=*/false, b,
      /*c_mod=*/(short)0, c, /*reuse_a=*/false, /*reuse_b=*/false);
}

// Async copy 16B global -> LDS (VGLOBAL GLOBAL_LOAD_ASYNC_TO_LDS_B128, GV
// addressing: VDST = LDS byte address, VADDR = 64-bit global address).
// Tracked by ASYNCcnt; completion enforced with s_wait_asynccnt.
static __device__ __forceinline__ void async_g2l_b128(unsigned lds_off,
                                                      const void* gptr) {
  asm volatile("global_load_async_to_lds_b128 %0, %1, off"
               :
               : "v"(lds_off), "v"(gptr)
               : "memory");
}

static __device__ __forceinline__ void wait_async0() {
  asm volatile("s_wait_asynccnt 0x0" ::: "memory");
}

__global__ void cast_f32_bf16(const float* __restrict__ src,
                              bf16_t* __restrict__ dst, int n) {
  int i = blockIdx.x * blockDim.x + threadIdx.x;
  if (i < n) dst[i] = (bf16_t)src[i];
}

// xp GEMM: M = B*T rows of x_bf16[.,64] times W_ih^T -> [M,512], + biases,
// stored bf16 transposed to [T][B][H] so the scan reads contiguous 16KB tiles.
// One wave computes one 16x16 output tile (K=64 -> 2 WMMA chunks).
__global__ __launch_bounds__(256) void xp_gemm(
    const bf16_t* __restrict__ xb, const bf16_t* __restrict__ wihb,
    const float* __restrict__ b_ih, const float* __restrict__ b_hh,
    bf16_t* __restrict__ xp) {
  const int wave = (blockIdx.x * 256 + threadIdx.x) >> 5;
  const int rt = wave >> 5;   // row tile (0..8191), 16 rows each
  const int ct = wave & 31;   // col tile (0..31),  16 cols each
  const int l = threadIdx.x & 31;
  const int half = l >> 4, lm = l & 15;

  const bf16_t* arow = xb + (size_t)(rt * 16 + lm) * D_ + half * 8;
  const bf16_t* brow = wihb + (size_t)(ct * 16 + lm) * D_ + half * 8;

  v8f acc = {};
#pragma unroll
  for (int c = 0; c < 2; ++c) {
    v16bf A = cat8(*(const v8bf*)(arow + c * 32),
                   *(const v8bf*)(arow + c * 32 + 16));
    v16bf Bf = cat8(*(const v8bf*)(brow + c * 32),
                    *(const v8bf*)(brow + c * 32 + 16));
    acc = wmma_bf16(A, Bf, acc);
  }

  const int n = ct * 16 + lm;
  const float bias = b_ih[n] + b_hh[n];
#pragma unroll
  for (int v = 0; v < 8; ++v) {
    int m = rt * 16 + v + 8 * half;  // global row in [B*T)
    int b = m >> 9;                  // m / T_
    int t = m & (T_ - 1);            // m % T_
    xp[((size_t)t * B_ + b) * H_ + n] = (bf16_t)(acc[v] + bias);
  }
}

// Persistent RNN scan. Grid: 16 blocks (one per 16-batch-row tile), 512 threads
// = 16 waves. Wave w owns output columns [w*32, w*32+32) (two 16-col WMMA
// tiles) of h_new, and row w of the fused W_ho dot product.
__global__ __launch_bounds__(512) void rnn_scan(
    const bf16_t* __restrict__ whhb, const bf16_t* __restrict__ xp,
    const float* __restrict__ w_ho, const float* __restrict__ b_ho,
    float* __restrict__ out) {
  __shared__ bf16_t hbuf[2][16 * H_];  // 32 KB, double-buffered hidden state
  __shared__ bf16_t xbuf[2][16 * H_];  // 32 KB, double-buffered xp_t tile

  const int tid = threadIdx.x;
  const int w = tid >> 5, l = tid & 31;
  const int half = l >> 4, lm = l & 15;
  const int b0 = blockIdx.x * 16;

  // h0 = 0
  for (int i = tid; i < 16 * H_; i += 512) hbuf[0][i] = (bf16_t)0.0f;

  // W_hh B-fragment row bases for this wave's two column tiles (constant in t)
  const int n0 = (w * 2) * 16 + lm;
  const int n1 = (w * 2 + 1) * 16 + lm;
  const bf16_t* wrow0 = whhb + (size_t)n0 * H_ + half * 8;
  const bf16_t* wrow1 = whhb + (size_t)n1 * H_ + half * 8;

  // w_ho cached in registers for the output phase (lane covers cols l+32*i)
  float wreg[16];
#pragma unroll
  for (int i = 0; i < 16; ++i) wreg[i] = w_ho[l + 32 * i];
  const float bho = b_ho[0];

  // Kick off async copy of the t=0 xp tile (16KB contiguous: 512 thr x 32B)
  {
    const char* g = (const char*)(xp + (size_t)b0 * H_) + tid * 32;
    unsigned lo = (unsigned)(uintptr_t)&xbuf[0][0] + tid * 32;
    async_g2l_b128(lo, g);
    async_g2l_b128(lo + 16, g + 16);
  }

  int p = 0;
  for (int t = 0; t < T_; ++t) {
    // xp tile for step t is in flight -> drain our ASYNCcnt, then make all
    // waves' copies (and last step's h stores) visible.
    wait_async0();
    __syncthreads();

    // Overlap: start streaming next step's xp tile while we do the WMMAs.
    if (t + 1 < T_) {
      const char* g =
          (const char*)(xp + ((size_t)(t + 1) * B_ + b0) * H_) + tid * 32;
      unsigned lo = (unsigned)(uintptr_t)&xbuf[p ^ 1][0] + tid * 32;
      async_g2l_b128(lo, g);
      async_g2l_b128(lo + 16, g + 16);
    }

    // This wave's xp values (C-fragment layout) from the staged LDS tile
    const bf16_t* xpt = &xbuf[p][0];
    float x0[8], x1[8];
#pragma unroll
    for (int v = 0; v < 8; ++v) {
      int mrow = v + 8 * half;
      x0[v] = (float)xpt[mrow * H_ + n0];
      x1[v] = (float)xpt[mrow * H_ + n1];
    }

    // h_new[:, wave cols] = h_old @ W_hh^T : 16 K-chunks of 32
    v8f acc0 = {}, acc1 = {};
    const bf16_t* hrow = &hbuf[p][lm * H_ + half * 8];
#pragma unroll
    for (int c = 0; c < 16; ++c) {
      v16bf A = cat8(*(const v8bf*)(hrow + c * 32),
                     *(const v8bf*)(hrow + c * 32 + 16));
      v16bf B0 = cat8(*(const v8bf*)(wrow0 + c * 32),
                      *(const v8bf*)(wrow0 + c * 32 + 16));
      acc0 = wmma_bf16(A, B0, acc0);
      v16bf B1 = cat8(*(const v8bf*)(wrow1 + c * 32),
                      *(const v8bf*)(wrow1 + c * 32 + 16));
      acc1 = wmma_bf16(A, B1, acc1);
    }

    // tanh + store new hidden state (bf16) to the other LDS buffer
    bf16_t* hn = hbuf[p ^ 1];
#pragma unroll
    for (int v = 0; v < 8; ++v) {
      int mrow = v + 8 * half;
      hn[mrow * H_ + n0] = (bf16_t)tanhf(acc0[v] + x0[v]);
      hn[mrow * H_ + n1] = (bf16_t)tanhf(acc1[v] + x1[v]);
    }

    __syncthreads();

    // Fused output: wave w reduces row w of h_new against w_ho
    float s = 0.f;
    const bf16_t* hr = &hn[w * H_];
#pragma unroll
    for (int i = 0; i < 16; ++i) s += (float)hr[l + 32 * i] * wreg[i];
#pragma unroll
    for (int m = 16; m >= 1; m >>= 1) s += __shfl_xor(s, m, 32);
    if (l == 0) out[(size_t)(b0 + w) * T_ + t] = s + bho;

    p ^= 1;
  }
}

extern "C" void kernel_launch(void* const* d_in, const int* in_sizes, int n_in,
                              void* d_out, int out_size, void* d_ws,
                              size_t ws_size, hipStream_t stream) {
  const float* x = (const float*)d_in[0];
  const float* W_ih = (const float*)d_in[1];
  const float* W_hh = (const float*)d_in[2];
  const float* b_ih = (const float*)d_in[3];
  const float* b_hh = (const float*)d_in[4];
  const float* W_ho = (const float*)d_in[5];
  const float* b_ho = (const float*)d_in[6];
  float* out = (float*)d_out;

  char* ws = (char*)d_ws;
  bf16_t* xb   = (bf16_t*)(ws);                                   // 16 MB
  bf16_t* wihb = (bf16_t*)(ws + (16u << 20));                     // 64 KB
  bf16_t* whhb = (bf16_t*)(ws + (16u << 20) + (64u << 10));       // 512 KB
  bf16_t* xp   = (bf16_t*)(ws + (16u << 20) + (64u << 10) + (512u << 10));

  const int nx = B_ * T_ * D_;   // 8,388,608
  const int nwih = H_ * D_;      // 32,768
  const int nwhh = H_ * H_;      // 262,144

  cast_f32_bf16<<<(nx + 255) / 256, 256, 0, stream>>>(x, xb, nx);
  cast_f32_bf16<<<(nwih + 255) / 256, 256, 0, stream>>>(W_ih, wihb, nwih);
  cast_f32_bf16<<<(nwhh + 255) / 256, 256, 0, stream>>>(W_hh, whhb, nwhh);

  // (B*T/16 row tiles) * (H/16 col tiles) = 262144 waves, 8 waves per block
  xp_gemm<<<32768, 256, 0, stream>>>(xb, wihb, b_ih, b_hh, xp);

  // One persistent block per 16-row batch tile; no inter-block sync needed.
  rnn_scan<<<B_ / 16, 512, 0, stream>>>(whhb, xp, W_ho, b_ho, out);
}
